// ModalityFusion_26680336843624
// MI455X (gfx1250) — matrix operand order repaired and checked by reading, not compiled, not run on hardware
//
#include <hip/hip_runtime.h>
#include <hip/hip_bf16.h>

typedef __attribute__((ext_vector_type(16))) _Float16 v16h;
typedef __attribute__((ext_vector_type(8)))  _Float16 v8h;
typedef __attribute__((ext_vector_type(8)))  float    v8f;

#define DIMC 384
#define NSP  1024   // H*W = 32*32
#define NBAT 32

static __device__ __forceinline__ v8f wmma_f16(v16h a, v16h b, v8f c) {
  // D = A(16x32 f16) x B(32x16 f16) + C(16x16 f32)
  return __builtin_amdgcn_wmma_f32_16x16x32_f16(false, a, false, b, (short)0, c, false, false);
}

// ---------------- conversion kernels ----------------

__global__ void cvt_w_kernel(const float* __restrict__ src, _Float16* __restrict__ dst, int n) {
  int i = blockIdx.x * blockDim.x + threadIdx.x;
  if (i < n) dst[i] = (_Float16)src[i];
}

// (b, c, n) f32 -> (b, n, c) f16
__global__ void transpose_cvt_kernel(const float* __restrict__ src, _Float16* __restrict__ dst) {
  int i = blockIdx.x * blockDim.x + threadIdx.x;
  int c = i % DIMC;
  int t = i / DIMC;
  int n = t % NSP;
  int b = t / NSP;
  dst[i] = (_Float16)src[(size_t)b * DIMC * NSP + (size_t)c * NSP + n];
}

// ---------------- projection kernel ----------------
// Qt(b,n,d) = Xr(b,n,:) . Wq(d,:)^T + bq   (stored f16, (n,d) row-major)
// Kt(b,n,d) likewise from Xm/Wk.
// V (b,d,n) from Xm/Wv (stored transposed for the attn (P x V^T) B-operand).
__global__ void __launch_bounds__(128)
proj_kernel(const _Float16* __restrict__ Xr, const _Float16* __restrict__ Xm,
            const _Float16* __restrict__ Wq, const _Float16* __restrict__ Wk,
            const _Float16* __restrict__ Wv,
            const float* __restrict__ bq, const float* __restrict__ bk,
            const float* __restrict__ bv,
            _Float16* __restrict__ Qt, _Float16* __restrict__ Kt,
            _Float16* __restrict__ V)
{
  const int wave  = threadIdx.x >> 5;
  const int lane  = threadIdx.x & 31;
  const int nn    = lane & 15;     // A row / B column / D column
  const int hi    = lane >> 4;     // lane-half
  const int batch = blockIdx.x >> 4;
  const int tile  = (blockIdx.x & 15) * 4 + wave;   // 0..63
  const int n0    = tile * 16;

  const size_t xbase = (size_t)batch * NSP * DIMC;
  union { v16h v; v8h h[2]; } u;
  v16h a[12];

  // ---- A fragments from Xr (rows n0..n0+15, full 384 K) ----
  #pragma unroll
  for (int kc = 0; kc < 12; ++kc) {
    const _Float16* p = Xr + xbase + (size_t)(n0 + nn) * DIMC + kc * 32;
    u.h[0] = *(const v8h*)(p + hi * 16);
    u.h[1] = *(const v8h*)(p + hi * 16 + 8);
    a[kc] = u.v;
  }
  // ---- Q projection ----
  for (int dt = 0; dt < 24; ++dt) {
    v8f acc = {};
    #pragma unroll
    for (int kc = 0; kc < 12; ++kc) {
      v16h bm = *(const v16h*)(Wq + (size_t)(dt * 16 + nn) * DIMC + kc * 32 + hi * 16);
      acc = wmma_f16(a[kc], bm, acc);
    }
    float bias = bq[dt * 16 + nn];
    #pragma unroll
    for (int r = 0; r < 8; ++r)
      Qt[xbase + (size_t)(n0 + hi * 8 + r) * DIMC + dt * 16 + nn] = (_Float16)(acc[r] + bias);
  }
  // ---- reload A from Xm ----
  #pragma unroll
  for (int kc = 0; kc < 12; ++kc) {
    const _Float16* p = Xm + xbase + (size_t)(n0 + nn) * DIMC + kc * 32;
    u.h[0] = *(const v8h*)(p + hi * 16);
    u.h[1] = *(const v8h*)(p + hi * 16 + 8);
    a[kc] = u.v;
  }
  // ---- K projection ----
  for (int dt = 0; dt < 24; ++dt) {
    v8f acc = {};
    #pragma unroll
    for (int kc = 0; kc < 12; ++kc) {
      v16h bm = *(const v16h*)(Wk + (size_t)(dt * 16 + nn) * DIMC + kc * 32 + hi * 16);
      acc = wmma_f16(a[kc], bm, acc);
    }
    float bias = bk[dt * 16 + nn];
    #pragma unroll
    for (int r = 0; r < 8; ++r)
      Kt[xbase + (size_t)(n0 + hi * 8 + r) * DIMC + dt * 16 + nn] = (_Float16)(acc[r] + bias);
  }
  // ---- V projection (stored (b, d, n)) ----
  for (int dt = 0; dt < 24; ++dt) {
    v8f acc = {};
    #pragma unroll
    for (int kc = 0; kc < 12; ++kc) {
      v16h bm = *(const v16h*)(Wv + (size_t)(dt * 16 + nn) * DIMC + kc * 32 + hi * 16);
      acc = wmma_f16(a[kc], bm, acc);
    }
    float bias = bv[dt * 16 + nn];
    #pragma unroll
    for (int r = 0; r < 8; ++r)
      V[(size_t)batch * DIMC * NSP + (size_t)(dt * 16 + nn) * NSP + (n0 + hi * 8 + r)]
          = (_Float16)(acc[r] + bias);
  }
}

// ---------------- attention kernel ----------------
// One wave per 16-query tile: S = scale * Qt_tile . K^T  (16 x 1024) -> LDS f16,
// softmax rows in LDS, then O = P . V^T -> Ot (b, n, c) f16.
__global__ void __launch_bounds__(64)
attn_kernel(const _Float16* __restrict__ Qt, const _Float16* __restrict__ Kt,
            const _Float16* __restrict__ V, _Float16* __restrict__ Ot)
{
  __shared__ _Float16 Sb[2][16][1024];   // 64 KB: per-wave 16x1024 score strip

  const int wave  = threadIdx.x >> 5;
  const int lane  = threadIdx.x & 31;
  const int nn    = lane & 15;
  const int hi    = lane >> 4;
  const int batch = blockIdx.x >> 5;
  const int tile  = (blockIdx.x & 31) * 2 + wave;   // 0..63
  const int n0    = tile * 16;
  const float scale = 0.051031036307982884f;        // 1/sqrt(384)

  const size_t qbase = (size_t)batch * NSP * DIMC;
  const size_t vbase = (size_t)batch * DIMC * NSP;
  union { v16h v; v8h h[2]; } u;
  v16h a[12];

  // A fragments: 16 query rows x full 384 K (f16)
  #pragma unroll
  for (int kc = 0; kc < 12; ++kc) {
    const _Float16* p = Qt + qbase + (size_t)(n0 + nn) * DIMC + kc * 32;
    u.h[0] = *(const v8h*)(p + hi * 16);
    u.h[1] = *(const v8h*)(p + hi * 16 + 8);
    a[kc] = u.v;
  }

  // ---- scores: 64 key tiles x 12 k-steps ----
  for (int mt = 0; mt < 64; ++mt) {
    if (mt + 1 < 64)
      __builtin_prefetch(Kt + qbase + (size_t)((mt + 1) * 16 + nn) * DIMC, 0, 3);
    v8f acc = {};
    #pragma unroll
    for (int kc = 0; kc < 12; ++kc) {
      v16h bm = *(const v16h*)(Kt + qbase + (size_t)(mt * 16 + nn) * DIMC + kc * 32 + hi * 16);
      acc = wmma_f16(a[kc], bm, acc);
    }
    #pragma unroll
    for (int r = 0; r < 8; ++r)
      Sb[wave][hi * 8 + r][mt * 16 + nn] = (_Float16)(acc[r] * scale);
  }
  __syncthreads();

  // ---- softmax: lane handles row nn, column-half hi (512 cols), rotated scan ----
  {
    const int r  = nn;
    const int cb = hi * 512;
    float mx = -3.0e38f;
    for (int i = 0; i < 512; ++i) {
      int c = cb + ((i + r * 8) & 511);
      mx = fmaxf(mx, (float)Sb[wave][r][c]);
    }
    mx = fmaxf(mx, __shfl_xor(mx, 16, 32));
    float sum = 0.0f;
    for (int i = 0; i < 512; ++i) {
      int c = cb + ((i + r * 8) & 511);
      sum += __expf((float)Sb[wave][r][c] - mx);
    }
    sum += __shfl_xor(sum, 16, 32);
    float inv = 1.0f / sum;
    for (int i = 0; i < 512; ++i) {
      int c = cb + ((i + r * 8) & 511);
      float p = __expf((float)Sb[wave][r][c] - mx) * inv;
      Sb[wave][r][c] = (_Float16)p;
    }
  }
  __syncthreads();

  // ---- O = P (16x1024) x V^T (1024x384) ----
  for (int ct = 0; ct < 24; ++ct) {
    v8f acc = {};
    for (int km = 0; km < 32; ++km) {
      const _Float16* p = &Sb[wave][nn][km * 32];
      u.h[0] = *(const v8h*)(p + hi * 16);
      u.h[1] = *(const v8h*)(p + hi * 16 + 8);
      v16h bm = *(const v16h*)(V + vbase + (size_t)(ct * 16 + nn) * NSP + km * 32 + hi * 16);
      acc = wmma_f16(u.v, bm, acc);
    }
    #pragma unroll
    for (int r = 0; r < 8; ++r)
      Ot[qbase + (size_t)(n0 + hi * 8 + r) * DIMC + ct * 16 + nn] = (_Float16)acc[r];
  }
}

// ---------------- final fuse kernel ----------------
// out(b,d,n) f32 = Wf(d,:) . O(:,n) + bf[d], computed as Ot rows x Wf^T.
__global__ void __launch_bounds__(128)
fuse_kernel(const _Float16* __restrict__ Ot, const _Float16* __restrict__ Wf,
            const float* __restrict__ bf, float* __restrict__ out)
{
  const int wave  = threadIdx.x >> 5;
  const int lane  = threadIdx.x & 31;
  const int nn    = lane & 15;
  const int hi    = lane >> 4;
  const int batch = blockIdx.x >> 4;
  const int tile  = (blockIdx.x & 15) * 4 + wave;
  const int n0    = tile * 16;

  const size_t obase = (size_t)batch * NSP * DIMC;
  union { v16h v; v8h h[2]; } u;
  v16h a[12];

  #pragma unroll
  for (int kc = 0; kc < 12; ++kc) {
    const _Float16* p = Ot + obase + (size_t)(n0 + nn) * DIMC + kc * 32;
    u.h[0] = *(const v8h*)(p + hi * 16);
    u.h[1] = *(const v8h*)(p + hi * 16 + 8);
    a[kc] = u.v;
  }
  for (int dt = 0; dt < 24; ++dt) {
    v8f acc = {};
    #pragma unroll
    for (int kc = 0; kc < 12; ++kc) {
      v16h bm = *(const v16h*)(Wf + (size_t)(dt * 16 + nn) * DIMC + kc * 32 + hi * 16);
      acc = wmma_f16(a[kc], bm, acc);
    }
    float bias = bf[dt * 16 + nn];
    #pragma unroll
    for (int r = 0; r < 8; ++r)
      out[(size_t)batch * DIMC * NSP + (size_t)(dt * 16 + nn) * NSP + (n0 + hi * 8 + r)]
          = acc[r] + bias;
  }
}

// ---------------- host launch ----------------

extern "C" void kernel_launch(void* const* d_in, const int* in_sizes, int n_in,
                              void* d_out, int out_size, void* d_ws, size_t ws_size,
                              hipStream_t stream) {
  const float* rgb = (const float*)d_in[0];
  const float* ms  = (const float*)d_in[1];
  const float* Wq  = (const float*)d_in[2];
  const float* bq  = (const float*)d_in[3];
  const float* Wk  = (const float*)d_in[4];
  const float* bk  = (const float*)d_in[5];
  const float* Wv  = (const float*)d_in[6];
  const float* bv  = (const float*)d_in[7];
  const float* Wf  = (const float*)d_in[8];
  const float* bf  = (const float*)d_in[9];
  float* out = (float*)d_out;

  char* ws = (char*)d_ws;
  const size_t big = (size_t)NBAT * NSP * DIMC * sizeof(_Float16);  // 24 MB
  const size_t wsz = (size_t)DIMC * DIMC * sizeof(_Float16);

  _Float16* Xr   = (_Float16*)ws; ws += big;
  _Float16* Xm   = (_Float16*)ws; ws += big;
  _Float16* Qt   = (_Float16*)ws; ws += big;
  _Float16* Kt   = (_Float16*)ws; ws += big;
  _Float16* Vd   = (_Float16*)ws; ws += big;
  _Float16* Ot   = (_Float16*)ws; ws += big;
  _Float16* Wq16 = (_Float16*)ws; ws += wsz;
  _Float16* Wk16 = (_Float16*)ws; ws += wsz;
  _Float16* Wv16 = (_Float16*)ws; ws += wsz;
  _Float16* Wf16 = (_Float16*)ws; ws += wsz;

  const int ne = NBAT * NSP * DIMC;           // 12,582,912 (multiple of 256)
  transpose_cvt_kernel<<<ne / 256, 256, 0, stream>>>(rgb, Xr);
  transpose_cvt_kernel<<<ne / 256, 256, 0, stream>>>(ms, Xm);
  const int nw = DIMC * DIMC;
  cvt_w_kernel<<<(nw + 255) / 256, 256, 0, stream>>>(Wq, Wq16, nw);
  cvt_w_kernel<<<(nw + 255) / 256, 256, 0, stream>>>(Wk, Wk16, nw);
  cvt_w_kernel<<<(nw + 255) / 256, 256, 0, stream>>>(Wv, Wv16, nw);
  cvt_w_kernel<<<(nw + 255) / 256, 256, 0, stream>>>(Wf, Wf16, nw);

  proj_kernel<<<NBAT * 16, 128, 0, stream>>>(Xr, Xm, Wq16, Wk16, Wv16,
                                             bq, bk, bv, Qt, Kt, Vd);
  attn_kernel<<<NBAT * 32, 64, 0, stream>>>(Qt, Kt, Vd, Ot);
  fuse_kernel<<<NBAT * 16, 128, 0, stream>>>(Ot, Wf16, bf, out);
}
